// SelfAttention_49185965474490
// MI455X (gfx1250) — matrix operand order repaired and checked by reading, not compiled, not run on hardware
//
#include <hip/hip_runtime.h>

// ---------------------------------------------------------------------------
// Self-attention forward for MI455X (gfx1250), wave32 + WMMA bf16.
//   B=2, S=2048, D=1024, H=16, HD=64.  ~103 GFLOP vs ~200MB HBM traffic
//   -> compute bound on matrix pipes.  All matmuls via
//   v_wmma_f32_16x16x32_bf16 (fp32 accum).  fp32->bf16 conversion is done
//   ONCE up front so GEMM inner loops are pure b128-load + wmma.
// ---------------------------------------------------------------------------

#define BQ   2
#define SQ   2048
#define DQ   1024
#define HQ   16
#define HDQ  64
#define BS   (BQ * SQ)      // 4096 rows

typedef __attribute__((ext_vector_type(16))) __bf16 v16bf;
typedef __attribute__((ext_vector_type(8)))  __bf16 v8bf;
typedef __attribute__((ext_vector_type(8)))  float  v8f;

static __device__ __forceinline__ __bf16 f2bf(float f) {
    unsigned u = __builtin_bit_cast(unsigned, f);
    unsigned r = u + 0x7FFFu + ((u >> 16) & 1u);   // round-to-nearest-even
    unsigned short h = (unsigned short)(r >> 16);
    return __builtin_bit_cast(__bf16, h);
}

static __device__ __forceinline__ v8f zero8() {
    v8f z;
#pragma unroll
    for (int i = 0; i < 8; ++i) z[i] = 0.f;
    return z;
}

static __device__ __forceinline__ v16bf cat8(v8bf lo, v8bf hi) {
    return __builtin_shufflevector(lo, hi, 0, 1, 2, 3, 4, 5, 6, 7,
                                           8, 9, 10, 11, 12, 13, 14, 15);
}

// 16 contiguous bf16 -> v16bf (two b128 loads)
static __device__ __forceinline__ v16bf ld16(const __bf16* p) {
    return cat8(*(const v8bf*)p, *(const v8bf*)(p + 8));
}
// A-operand gather: elements 0..7 = p0[0..7], 8..15 = p1[0..7]
static __device__ __forceinline__ v16bf ld8x2(const __bf16* p0, const __bf16* p1) {
    return cat8(*(const v8bf*)p0, *(const v8bf*)p1);
}

static __device__ __forceinline__ float rmax16(float v) {
    v = fmaxf(v, __shfl_xor(v, 1, 32));
    v = fmaxf(v, __shfl_xor(v, 2, 32));
    v = fmaxf(v, __shfl_xor(v, 4, 32));
    v = fmaxf(v, __shfl_xor(v, 8, 32));
    return v;
}
static __device__ __forceinline__ float rsum16(float v) {
    v += __shfl_xor(v, 1, 32);
    v += __shfl_xor(v, 2, 32);
    v += __shfl_xor(v, 4, 32);
    v += __shfl_xor(v, 8, 32);
    return v;
}

// ---------------------------------------------------------------------------
// One-pass fp32 -> bf16 conversion (memory bound; 8 elems/thread).
// ---------------------------------------------------------------------------
__global__ __launch_bounds__(256) void cvt_kernel(
    const float* __restrict__ in, __bf16* __restrict__ out)
{
    const size_t i = ((size_t)blockIdx.x * blockDim.x + threadIdx.x) * 8;
    float4 a = *(const float4*)(in + i);
    float4 b = *(const float4*)(in + i + 4);
    v8bf o;
    o[0] = f2bf(a.x); o[1] = f2bf(a.y); o[2] = f2bf(a.z); o[3] = f2bf(a.w);
    o[4] = f2bf(b.x); o[5] = f2bf(b.y); o[6] = f2bf(b.z); o[7] = f2bf(b.w);
    *(v8bf*)(out + i) = o;
}

// ---------------------------------------------------------------------------
// Projection: out = x @ W^T (+RoPE for Q/K).  One wave = 32 rows x 64 cols
// (two A tiles share every B load -> 8 WMMAs per k-step).
// mode 0: Q -> rope -> [b][h][s][64]
// mode 1: K -> rope -> [b][h][s][64]
// mode 2: V ->          [b][h][64][s]   (transposed for PV B-operand)
// ---------------------------------------------------------------------------
__global__ __launch_bounds__(128) void proj_kernel(
    const __bf16* __restrict__ x,    // [BS][DQ] bf16
    const __bf16* __restrict__ w,    // [DQ][DQ] bf16
    const float* __restrict__ fcos,  // [SQ][32]
    const float* __restrict__ fsin,  // [SQ][32]
    __bf16* __restrict__ out, int mode)
{
    const int wid  = blockIdx.x * (blockDim.x >> 5) + (threadIdx.x >> 5);
    const int lane = threadIdx.x & 31;
    const int l15  = lane & 15;
    const int hl   = lane >> 4;
    const int mt   = wid & 127;   // 32-row tile of BS
    const int ns   = wid >> 7;    // 64-col strip (0..15)
    const int n0   = ns * 64;

    v8f acc[2][4];
#pragma unroll
    for (int t = 0; t < 2; ++t)
#pragma unroll
        for (int i = 0; i < 4; ++i) acc[t][i] = zero8();

    const __bf16* arow0 = x + (size_t)(mt * 32 + l15) * DQ;
    const __bf16* arow1 = arow0 + (size_t)16 * DQ;

    for (int k0 = 0; k0 < DQ; k0 += 32) {
        v16bf a0 = ld8x2(arow0 + k0 + 16 * hl, arow0 + k0 + 8 + 16 * hl);
        v16bf a1 = ld8x2(arow1 + k0 + 16 * hl, arow1 + k0 + 8 + 16 * hl);
#pragma unroll
        for (int nb = 0; nb < 4; ++nb) {
            const int col = n0 + nb * 16 + l15;
            v16bf b = ld16(w + (size_t)col * DQ + k0 + 16 * hl);
            acc[0][nb] = __builtin_amdgcn_wmma_f32_16x16x32_bf16(
                false, a0, false, b, (short)0, acc[0][nb], false, false);
            acc[1][nb] = __builtin_amdgcn_wmma_f32_16x16x32_bf16(
                false, a1, false, b, (short)0, acc[1][nb], false, false);
        }
    }

    // Epilogue: C/D layout -> lane l holds col n0+nb*16+l15, rows r+8*hl.
#pragma unroll
    for (int t = 0; t < 2; ++t) {
        const int rbase = mt * 32 + t * 16 + 8 * hl;
#pragma unroll
        for (int nb = 0; nb < 4; ++nb) {
            const int c  = n0 + nb * 16 + l15;
            const int h  = c >> 6;
            const int hd = c & 63;
#pragma unroll
            for (int r = 0; r < 8; ++r) {
                const int m  = rbase + r;           // global BS row
                const int bb = m >> 11;             // / SQ
                const int s  = m & (SQ - 1);
                float val = acc[t][nb][r];
                if (mode < 2) {
                    // RoPE: partner element hd^1 lives in lane^1 (same r,nb).
                    float partner = __shfl_xor(val, 1, 32);
                    const int i = hd >> 1;
                    const float cv = fcos[s * 32 + i];
                    const float sv = fsin[s * 32 + i];
                    val = (hd & 1) ? (partner * sv + val * cv)
                                   : (val * cv - partner * sv);
                    out[(((size_t)(bb * HQ + h) * SQ + s) << 6) + hd] = f2bf(val);
                } else {
                    out[((size_t)(bb * HQ + h) * HDQ + hd) * SQ + s] = f2bf(val);
                }
            }
        }
    }
}

// ---------------------------------------------------------------------------
// Flash attention: one wave per (b,h, 16-query tile); online softmax over
// 32-key chunks; causal mask.  8 WMMAs per 32 keys.
// ---------------------------------------------------------------------------
__global__ __launch_bounds__(128) void attn_kernel(
    const __bf16* __restrict__ qh,   // [b][h][s][64]
    const __bf16* __restrict__ kh,   // [b][h][s][64]
    const __bf16* __restrict__ vT,   // [b][h][64][s]
    __bf16* __restrict__ aout)       // [BS][DQ]
{
    __shared__ __align__(16) __bf16 lds_p[4][16][32];  // per-wave P tile
    const int wslot = threadIdx.x >> 5;
    const int wid   = blockIdx.x * 4 + wslot;
    const int lane  = threadIdx.x & 31;
    const int l15   = lane & 15;
    const int hl    = lane >> 4;

    const int qt = wid & 127;           // S/16
    const int bh = wid >> 7;            // b*H + h
    const int q0 = qt * 16;

    const __bf16* qbase = qh + (size_t)bh * SQ * 64;
    const __bf16* kbase = kh + (size_t)bh * SQ * 64;
    const __bf16* vbase = vT + (size_t)bh * 64 * SQ;

    // Q A-operands for dims 0..31 and 32..63, held for the whole loop.
    const __bf16* qp = qbase + (size_t)(q0 + l15) * 64;
    const v16bf Alo = ld8x2(qp + 16 * hl, qp + 8 + 16 * hl);
    const v16bf Ahi = ld8x2(qp + 32 + 16 * hl, qp + 40 + 16 * hl);

    v8f o[4];
#pragma unroll
    for (int i = 0; i < 4; ++i) o[i] = zero8();
    float mrow[8], srow[8];
#pragma unroll
    for (int r = 0; r < 8; ++r) { mrow[r] = -3.0e38f; srow[r] = 0.f; }

    const int jmax = (q0 + 47) >> 5;    // covers keys 0 .. q0+15
    for (int j = 0; j < jmax; ++j) {
        const int n0 = j << 5;

        // ---- S = Q K^T for two 16-key tiles (4 WMMAs) ----
        v8f sc[2];
#pragma unroll
        for (int t = 0; t < 2; ++t) {
            const __bf16* kp = kbase + (size_t)(n0 + 16 * t + l15) * 64;
            v16bf B0 = ld16(kp + 16 * hl);
            v16bf B1 = ld16(kp + 32 + 16 * hl);
            v8f z = zero8();
            z = __builtin_amdgcn_wmma_f32_16x16x32_bf16(false, Alo, false, B0,
                                                        (short)0, z, false, false);
            z = __builtin_amdgcn_wmma_f32_16x16x32_bf16(false, Ahi, false, B1,
                                                        (short)0, z, false, false);
            sc[t] = z;
        }

        // ---- scale + causal mask (wave-uniform diagonal test) ----
        const bool diag = (n0 + 31 > q0);
#pragma unroll
        for (int t = 0; t < 2; ++t) {
#pragma unroll
            for (int r = 0; r < 8; ++r) {
                float v = sc[t][r] * 0.125f;    // 1/sqrt(64)
                if (diag) {
                    const int key = n0 + 16 * t + l15;
                    const int qq  = q0 + 8 * hl + r;
                    if (key > qq) v = -3.0e38f;
                }
                sc[t][r] = v;
            }
        }

        // ---- online softmax: rows live across 16-lane half-waves ----
#pragma unroll
        for (int r = 0; r < 8; ++r) {
            float mx = rmax16(fmaxf(sc[0][r], sc[1][r]));
            const float mnew  = fmaxf(mrow[r], mx);
            const float alpha = __expf(mrow[r] - mnew);
            float p0 = __expf(sc[0][r] - mnew);
            float p1 = __expf(sc[1][r] - mnew);
            sc[0][r] = p0; sc[1][r] = p1;
            srow[r] = srow[r] * alpha + rsum16(p0 + p1);
            mrow[r] = mnew;
#pragma unroll
            for (int nb = 0; nb < 4; ++nb) o[nb][r] *= alpha;
        }

        // ---- P tile: C-layout -> A-layout via private LDS region ----
#pragma unroll
        for (int t = 0; t < 2; ++t)
#pragma unroll
            for (int r = 0; r < 8; ++r)
                lds_p[wslot][8 * hl + r][16 * t + l15] = f2bf(sc[t][r]);
        asm volatile("s_wait_dscnt 0" ::: "memory");
        const __bf16* pr = &lds_p[wslot][l15][0];
        v16bf pA = ld8x2(pr + 16 * hl, pr + 8 + 16 * hl);

        // ---- O += P V (4 WMMAs against transposed V) ----
#pragma unroll
        for (int nb = 0; nb < 4; ++nb) {
            const __bf16* vp = vbase + (size_t)(nb * 16 + l15) * SQ + n0 + 16 * hl;
            v16bf Bv = ld16(vp);
            o[nb] = __builtin_amdgcn_wmma_f32_16x16x32_bf16(
                false, pA, false, Bv, (short)0, o[nb], false, false);
        }
    }

    // ---- normalize and store bf16 [b][s][d] for the output projection ----
    const int b = bh >> 4, h = bh & 15;
#pragma unroll
    for (int r = 0; r < 8; ++r) {
        const float inv = 1.0f / srow[r];
        const int sq = q0 + 8 * hl + r;
        const size_t rowoff = ((size_t)b * SQ + sq) * DQ + h * 64;
#pragma unroll
        for (int nb = 0; nb < 4; ++nb)
            aout[rowoff + nb * 16 + l15] = f2bf(o[nb][r] * inv);
    }
}

// ---------------------------------------------------------------------------
// Output projection: d_out = attn_out @ wo^T.  32 rows x 64 cols per wave.
// ---------------------------------------------------------------------------
__global__ __launch_bounds__(128) void oproj_kernel(
    const __bf16* __restrict__ a,   // [BS][DQ] bf16
    const __bf16* __restrict__ w,   // wo [DQ][DQ] bf16
    float* __restrict__ out)        // [BS][DQ]
{
    const int wid  = blockIdx.x * (blockDim.x >> 5) + (threadIdx.x >> 5);
    const int lane = threadIdx.x & 31;
    const int l15  = lane & 15;
    const int hl   = lane >> 4;
    const int mt   = wid & 127;
    const int ns   = wid >> 7;
    const int n0   = ns * 64;

    v8f acc[2][4];
#pragma unroll
    for (int t = 0; t < 2; ++t)
#pragma unroll
        for (int i = 0; i < 4; ++i) acc[t][i] = zero8();

    const __bf16* arow0 = a + (size_t)(mt * 32 + l15) * DQ;
    const __bf16* arow1 = arow0 + (size_t)16 * DQ;

    for (int k0 = 0; k0 < DQ; k0 += 32) {
        v16bf a0 = ld8x2(arow0 + k0 + 16 * hl, arow0 + k0 + 8 + 16 * hl);
        v16bf a1 = ld8x2(arow1 + k0 + 16 * hl, arow1 + k0 + 8 + 16 * hl);
#pragma unroll
        for (int nb = 0; nb < 4; ++nb) {
            const int col = n0 + nb * 16 + l15;
            v16bf b = ld16(w + (size_t)col * DQ + k0 + 16 * hl);
            acc[0][nb] = __builtin_amdgcn_wmma_f32_16x16x32_bf16(
                false, a0, false, b, (short)0, acc[0][nb], false, false);
            acc[1][nb] = __builtin_amdgcn_wmma_f32_16x16x32_bf16(
                false, a1, false, b, (short)0, acc[1][nb], false, false);
        }
    }

#pragma unroll
    for (int t = 0; t < 2; ++t) {
        const int rbase = mt * 32 + t * 16 + 8 * hl;
#pragma unroll
        for (int nb = 0; nb < 4; ++nb) {
            const int c = n0 + nb * 16 + l15;
#pragma unroll
            for (int r = 0; r < 8; ++r)
                out[(size_t)(rbase + r) * DQ + c] = acc[t][nb][r];
        }
    }
}

// ---------------------------------------------------------------------------
extern "C" void kernel_launch(void* const* d_in, const int* in_sizes, int n_in,
                              void* d_out, int out_size, void* d_ws, size_t ws_size,
                              hipStream_t stream)
{
    (void)in_sizes; (void)n_in; (void)out_size; (void)ws_size;
    const float* x  = (const float*)d_in[0];
    const float* wq = (const float*)d_in[1];
    const float* wk = (const float*)d_in[2];
    const float* wv = (const float*)d_in[3];
    const float* wo = (const float*)d_in[4];
    const float* fc = (const float*)d_in[5];
    const float* fs = (const float*)d_in[6];

    char* ws = (char*)d_ws;                       // 48 MiB used
    __bf16* qh  = (__bf16*)(ws);                  // 8 MiB [b][h][s][64]
    __bf16* kh  = (__bf16*)(ws + (8u  << 20));    // 8 MiB [b][h][s][64]
    __bf16* vT  = (__bf16*)(ws + (16u << 20));    // 8 MiB [b][h][64][s]
    __bf16* ao  = (__bf16*)(ws + (24u << 20));    // 8 MiB [BS][DQ]
    __bf16* xbf = (__bf16*)(ws + (32u << 20));    // 8 MiB [BS][DQ]
    __bf16* wqb = (__bf16*)(ws + (40u << 20));    // 2 MiB
    __bf16* wkb = (__bf16*)(ws + (42u << 20));    // 2 MiB
    __bf16* wvb = (__bf16*)(ws + (44u << 20));    // 2 MiB
    __bf16* wob = (__bf16*)(ws + (46u << 20));    // 2 MiB

    // fp32 -> bf16 staging (memory bound, one pass)
    cvt_kernel<<<dim3((BS * DQ) / (256 * 8)), dim3(256), 0, stream>>>(x,  xbf);
    cvt_kernel<<<dim3((DQ * DQ) / (256 * 8)), dim3(256), 0, stream>>>(wq, wqb);
    cvt_kernel<<<dim3((DQ * DQ) / (256 * 8)), dim3(256), 0, stream>>>(wk, wkb);
    cvt_kernel<<<dim3((DQ * DQ) / (256 * 8)), dim3(256), 0, stream>>>(wv, wvb);
    cvt_kernel<<<dim3((DQ * DQ) / (256 * 8)), dim3(256), 0, stream>>>(wo, wob);

    const dim3 blk(128);
    // projections: (BS/32) * (DQ/64) = 2048 waves -> 512 blocks
    proj_kernel<<<dim3(512), blk, 0, stream>>>(xbf, wqb, fc, fs, qh, 0);
    proj_kernel<<<dim3(512), blk, 0, stream>>>(xbf, wkb, fc, fs, kh, 1);
    proj_kernel<<<dim3(512), blk, 0, stream>>>(xbf, wvb, fc, fs, vT, 2);
    // attention: B*H*(S/16) = 4096 waves -> 1024 blocks
    attn_kernel<<<dim3(1024), blk, 0, stream>>>(qh, kh, vT, ao);
    // output projection
    oproj_kernel<<<dim3(512), blk, 0, stream>>>(ao, wob, (float*)d_out);
}
